// Block_56178172232284
// MI455X (gfx1250) — matrix operand order repaired and checked by reading, not compiled, not run on hardware
//
#include <hip/hip_runtime.h>
#include <hip/hip_bf16.h>
#include <math.h>

typedef __attribute__((ext_vector_type(16))) _Float16 v16h;
typedef __attribute__((ext_vector_type(8)))  _Float16 v8h;
typedef __attribute__((ext_vector_type(8)))  float    v8f;

#define TT   32768   // B*N tokens
#define NN   4096    // tokens per image
#define BB   8
#define CC   128
#define HH   4
#define HD   32
#define WSZ  64      // spatial W (=H)
#define PL   64      // pooled length
#define HID  341
#define HID2 682
#define HIDP 352     // HID padded to multiple of 32 for WMMA K
#define UP   704     // fc1 output cols padded to multiple of 64 (44 n-tiles)
#define LOC  9

// ---------------- wave32 helpers ----------------
__device__ inline float wsum(float v){
#pragma unroll
  for (int o = 16; o; o >>= 1) v += __shfl_xor(v, o, 32);
  return v;
}
__device__ inline float wmax32(float v){
#pragma unroll
  for (int o = 16; o; o >>= 1) v = fmaxf(v, __shfl_xor(v, o, 32));
  return v;
}
__device__ inline float gelu_exact(float x){
  return 0.5f * x * (1.f + erff(x * 0.70710678118654752f));
}

// ---------------- generic WMMA GEMM: out[m,n] = sum_k A[m,k]*Wm[n,k] (+epilogue) ----------------
// All B rows touched by the tile grid must be readable (weights pre-padded with zeros);
// the WMMA main loop is completely branch-free (EXEC stays all-ones). All fragment loads of a
// k-step are issued before the WMMA sequence so the scheduler can use partial s_wait_loadcnt.
// EPI: 0 plain f32  1 +bias  2 +bias,gelu  3 +bias+res  4 +pb gather (attn_pool)  5 x_p: +xl, f16 out
template<int EPI, int NT>
__global__ __launch_bounds__(256) void gemm_wmma(
    const _Float16* __restrict__ A, const _Float16* __restrict__ Wm,
    const float* __restrict__ bias, const float* __restrict__ res,
    float* __restrict__ out, _Float16* __restrict__ outh,
    int M, int K, int Nd, int ldA, int ldW, int ldO,
    long long sA, long long sW, long long sO,
    const int* __restrict__ relidx, const float* __restrict__ pbt)
{
  const int lane  = threadIdx.x & 31;
  const int wave  = threadIdx.x >> 5;
  const int mtile = blockIdx.x * 8 + wave;
  if (mtile * 16 >= M) return;               // wave-uniform
  const int ng0   = blockIdx.y * NT;
  const int batch = blockIdx.z;
  A  += (long long)batch * sA;
  Wm += (long long)batch * sW;
  const int rsel = lane & 15;
  const int hsel = lane >> 4;

  v8f acc[NT];
#pragma unroll
  for (int t = 0; t < NT; t++)
#pragma unroll
    for (int i = 0; i < 8; i++) acc[t][i] = 0.f;

  const _Float16* arow = A + (long long)(mtile * 16 + rsel) * ldA;
  const _Float16* wrow[NT];
#pragma unroll
  for (int t = 0; t < NT; t++)
    wrow[t] = Wm + (long long)((ng0 + t) * 16 + rsel) * ldW;

  for (int kk = 0; kk < K; kk += 32){
    // issue ALL loads of this k-step first, then the WMMA chain (enables partial loadcnt waits)
    v16h a;
    {
      v8h lo = *(const v8h*)(arow + kk + hsel * 8);
      v8h hi = *(const v8h*)(arow + kk + 16 + hsel * 8);
#pragma unroll
      for (int i = 0; i < 8; i++){ a[i] = lo[i]; a[i + 8] = hi[i]; }
    }
    v16h b[NT];
#pragma unroll
    for (int t = 0; t < NT; t++){
      v8h lo = *(const v8h*)(wrow[t] + kk + hsel * 8);
      v8h hi = *(const v8h*)(wrow[t] + kk + 16 + hsel * 8);
#pragma unroll
      for (int i = 0; i < 8; i++){ b[t][i] = lo[i]; b[t][i + 8] = hi[i]; }
    }
#pragma unroll
    for (int t = 0; t < NT; t++)
      acc[t] = __builtin_amdgcn_wmma_f32_16x16x32_f16(
                 false, a, false, b[t], (short)0, acc[t], false, false);
  }

  // epilogue: C/D layout — VGPR j, lanes0-15: M=j, lanes16-31: M=8+j; N=lane&15
#pragma unroll
  for (int t = 0; t < NT; t++){
    int ncol = (ng0 + t) * 16 + rsel;
    bool nv = (ncol < Nd);                    // only ragged for fc1 (stores/bias)
#pragma unroll
    for (int j = 0; j < 8; j++){
      int  mrow = mtile * 16 + j + 8 * hsel;
      float v = acc[t][j];
      if constexpr (EPI == 0){
        if (nv) out[(long long)batch * sO + (long long)mrow * ldO + ncol] = v;
      } else if constexpr (EPI == 1){
        if (nv) out[(long long)mrow * ldO + ncol] = v + bias[ncol];
      } else if constexpr (EPI == 2){
        if (nv) out[(long long)mrow * ldO + ncol] = gelu_exact(v + bias[ncol]);
      } else if constexpr (EPI == 3){
        if (nv){ long long ix = (long long)mrow * ldO + ncol;
                 out[ix] = v + bias[ncol] + res[ix]; }
      } else if constexpr (EPI == 4){
        if (nv){
          int hh = batch & 3;
          int idx = relidx[(long long)mrow * PL + ncol];
          out[(long long)batch * sO + (long long)mrow * ldO + ncol] = v + pbt[hh * 1024 + idx];
        }
      } else { // EPI == 5 : x_p, add x_l, write f16 proj input at [b,n, h*32+d]
        if (nv){
          int bb = batch >> 2, hh = batch & 3;
          long long ix = ((long long)bb * NN + mrow) * CC + hh * HD + ncol;
          outh[ix] = (_Float16)(v + res[ix]);
        }
      }
    }
  }
}

// ---------------- LayerNorm over 128 cols, one wave per row, f16 output ----------------
__global__ __launch_bounds__(256) void ln_k(const float* __restrict__ in,
    const float* __restrict__ g, const float* __restrict__ b,
    _Float16* __restrict__ out, int rows)
{
  int row = blockIdx.x * 8 + (threadIdx.x >> 5);
  if (row >= rows) return;
  int lane = threadIdx.x & 31;
  const float4 v = ((const float4*)(in + (long long)row * CC))[lane];
  float m = wsum(v.x + v.y + v.z + v.w) * (1.f / 128.f);
  float a0 = v.x - m, a1 = v.y - m, a2 = v.z - m, a3 = v.w - m;
  float var = wsum(a0*a0 + a1*a1 + a2*a2 + a3*a3) * (1.f / 128.f);
  float rs = rsqrtf(var + 1e-5f);
  int c = lane * 4;
  _Float16* o = out + (long long)row * CC + c;
  o[0] = (_Float16)(a0 * rs * g[c + 0] + b[c + 0]);
  o[1] = (_Float16)(a1 * rs * g[c + 1] + b[c + 1]);
  o[2] = (_Float16)(a2 * rs * g[c + 2] + b[c + 2]);
  o[3] = (_Float16)(a3 * rs * g[c + 3] + b[c + 3]);
}

// ---------------- q: l2-normalize per (token,head), build q_s ----------------
__global__ __launch_bounds__(256) void qprep_k(const float* __restrict__ q,
    const float* __restrict__ temp, const float* __restrict__ qe,
    const float* __restrict__ sls, float* __restrict__ qn, _Float16* __restrict__ qs)
{
  int w = blockIdx.x * 8 + (threadIdx.x >> 5);   // (t,h)
  int lane = threadIdx.x & 31;
  int t = w >> 2, h = w & 3;
  float v = q[(long long)t * CC + h * HD + lane];
  float nrm = sqrtf(wsum(v * v));
  float qv = v / fmaxf(nrm, 1e-12f);
  qn[(long long)t * CC + h * HD + lane] = qv;
  float sc = log1pf(expf(temp[h])) * sls[0];     // softplus(temp) * seq_length_scale
  int b = t >> 12, n = t & (NN - 1);
  qs[(((long long)(b * HH + h)) * NN + n) * HD + lane] = (_Float16)((qv + qe[h * HD + lane]) * sc);
}

// ---------------- k (local): l2-normalize per (token,head) ----------------
__global__ __launch_bounds__(256) void kprep_k(const float* __restrict__ kv, float* __restrict__ kl)
{
  int w = blockIdx.x * 8 + (threadIdx.x >> 5);
  int lane = threadIdx.x & 31;
  int t = w >> 2, h = w & 3;
  float v = kv[(long long)t * 256 + h * HD + lane];
  float nrm = sqrtf(wsum(v * v));
  kl[(long long)t * CC + h * HD + lane] = v / fmaxf(nrm, 1e-12f);
}

// ---------------- 8x8 average pool over spatial ----------------
__global__ __launch_bounds__(128) void pool_k(const float* __restrict__ xi, float* __restrict__ pooled)
{
  int bp = blockIdx.x;           // 0..511 (b*64+pl)
  int c = threadIdx.x;
  int b = bp >> 6, pl = bp & 63;
  int py = pl >> 3, px = pl & 7;
  float s = 0.f;
  for (int sy = 0; sy < 8; sy++)
    for (int sx = 0; sx < 8; sx++)
      s += xi[(long long)(b * NN + (py * 8 + sy) * WSZ + (px * 8 + sx)) * CC + c];
  pooled[(long long)bp * CC + c] = s * (1.f / 64.f);
}

// ---------------- pooled kv: normalize k_p, transpose v_p ----------------
__global__ __launch_bounds__(256) void kpprep_k(const float* __restrict__ kvp,
    _Float16* __restrict__ kp, _Float16* __restrict__ vpt)
{
  int w = blockIdx.x * 8 + (threadIdx.x >> 5);   // (bp,h), 2048 waves
  int lane = threadIdx.x & 31;
  int bp = w >> 2, h = w & 3;
  int b = bp >> 6, pl = bp & 63;
  float kvv = kvp[(long long)bp * 256 + h * HD + lane];
  float nrm = sqrtf(wsum(kvv * kvv));
  kp[(((long long)(b * HH + h)) * PL + pl) * HD + lane] = (_Float16)(kvv / fmaxf(nrm, 1e-12f));
  float vv = kvp[(long long)bp * 256 + CC + h * HD + lane];
  vpt[(((long long)(b * HH + h)) * HD + lane) * PL + pl] = (_Float16)vv;
}

// ---------------- CPB bias MLP: relu(table@w1^T+b1)@w2^T+b2 -> pbt[4][1024] ----------------
__global__ __launch_bounds__(256) void cpb_k(const float* __restrict__ tab,
    const float* __restrict__ w1, const float* __restrict__ b1,
    const float* __restrict__ w2, const float* __restrict__ b2, float* __restrict__ pbt)
{
  __shared__ float hid[512];
  __shared__ float red[256];
  int r = blockIdx.x, tid = threadIdx.x;
  float c0 = tab[r * 2 + 0], c1 = tab[r * 2 + 1];
  for (int j = tid; j < 512; j += 256)
    hid[j] = fmaxf(c0 * w1[j * 2 + 0] + c1 * w1[j * 2 + 1] + b1[j], 0.f);
  __syncthreads();
  for (int h = 0; h < 4; h++){
    red[tid] = hid[tid] * w2[h * 512 + tid] + hid[tid + 256] * w2[h * 512 + tid + 256];
    __syncthreads();
    for (int o = 128; o; o >>= 1){ if (tid < o) red[tid] += red[tid + o]; __syncthreads(); }
    if (tid == 0) pbt[h * 1024 + r] = red[0] + b2[h];
    __syncthreads();
  }
}

// ---------------- fused local attention + 73-way softmax + x_l + a_p ----------------
__global__ __launch_bounds__(256) void attn_k(
    const _Float16* __restrict__ qs, const float* __restrict__ qn,
    const float* __restrict__ kl, const float* __restrict__ kv,
    const float* __restrict__ AP, const float* __restrict__ rbl,
    const float* __restrict__ lt, const float* __restrict__ lb,
    const unsigned char* __restrict__ mask,
    float* __restrict__ xl, _Float16* __restrict__ ap)
{
  int w = blockIdx.x * 8 + (threadIdx.x >> 5);   // (b,h,n)
  int lane = threadIdx.x & 31;                   // lane = d
  int n = w & (NN - 1);
  int h = (w >> 12) & 3;
  int b = w >> 14;
  int y = n >> 6, x = n & 63;
  long long bh = (long long)(b * HH + h);
  float qsv = (float)qs[(bh * NN + n) * HD + lane];
  float qnv = qn[((long long)(b * NN + n)) * CC + h * HD + lane];
  float llog[LOC], ldot[LOC];
  int tok[LOC];
#pragma unroll
  for (int l = 0; l < LOC; l++){
    int dy = l / 3 - 1, dx = l % 3 - 1;
    int ny = y + dy, nx = x + dx;
    bool ok = (ny >= 0 && ny < WSZ && nx >= 0 && nx < WSZ);
    int tk = ok ? (b * NN + ny * WSZ + nx) : -1;
    tok[l] = tk;
    float kvv = ok ? kl[((long long)tk) * CC + h * HD + lane] : 0.f;
    float lg = wsum(qsv * kvv) + rbl[h * LOC + l];          // replicated
    if (mask[((long long)(b * NN + n)) * LOC + l]) lg = -INFINITY;
    llog[l] = lg;
    ldot[l] = wsum(qnv * lt[(h * HD + lane) * LOC + l]);    // replicated
  }
  const float* apr = AP + (bh * NN + n) * PL;
  float p0 = apr[lane], p1 = apr[lane + 32];
  float mx = wmax32(fmaxf(p0, p1));
#pragma unroll
  for (int l = 0; l < LOC; l++) mx = fmaxf(mx, llog[l]);
  float sl = 0.f;
#pragma unroll
  for (int l = 0; l < LOC; l++) sl += expf(llog[l] - mx);
  float e0 = expf(p0 - mx), e1 = expf(p1 - mx);
  float inv = 1.f / (sl + wsum(e0 + e1));
  ap[(bh * NN + n) * PL + lane]      = (_Float16)(e0 * inv);
  ap[(bh * NN + n) * PL + lane + 32] = (_Float16)(e1 * inv);
  float acc = 0.f;
#pragma unroll
  for (int l = 0; l < LOC; l++){
    float lm = ldot[l] + lb[h * LOC + l] + expf(llog[l] - mx) * inv;   // l_mix
    float vv = (tok[l] >= 0) ? kv[((long long)tok[l]) * 256 + CC + h * HD + lane] : 0.f;
    acc += lm * vv;
  }
  xl[((long long)(b * NN + n)) * CC + h * HD + lane] = acc;
}

// ---------------- depthwise 3x3 conv + GELU gate -> f16 (K padded to 352) ----------------
__global__ __launch_bounds__(352) void dwgate_k(const float* __restrict__ u,
    const float* __restrict__ dww, const float* __restrict__ dwb, _Float16* __restrict__ mh)
{
  int t = blockIdx.x;
  int ch = threadIdx.x;
  if (ch >= HIDP) return;
  if (ch >= HID){ mh[(long long)t * HIDP + ch] = (_Float16)0.f; return; }
  int b = t >> 12, n = t & (NN - 1);
  int y = n >> 6, x = n & 63;
  float s = dwb[ch];
  for (int ky = 0; ky < 3; ky++){
    int ny = y + ky - 1;
    if (ny < 0 || ny >= WSZ) continue;
    for (int kx = 0; kx < 3; kx++){
      int nx = x + kx - 1;
      if (nx < 0 || nx >= WSZ) continue;
      s += u[(long long)(b * NN + ny * WSZ + nx) * UP + ch] * dww[ch * 9 + ky * 3 + kx];
    }
  }
  float v1 = u[(long long)t * UP + HID + ch];
  mh[(long long)t * HIDP + ch] = (_Float16)(gelu_exact(s) * v1);
}

// ---------------- f32 -> f16 weight convert with row AND column zero-padding ----------------
__global__ __launch_bounds__(256) void cvt_k(const float* __restrict__ src,
    _Float16* __restrict__ dst, int src_rows, int dst_rows, int sc, int dc)
{
  long long i = (long long)blockIdx.x * 256 + threadIdx.x;
  if (i >= (long long)dst_rows * dc) return;
  int r = (int)(i / dc), c = (int)(i % dc);
  dst[i] = (r < src_rows && c < sc) ? (_Float16)src[(long long)r * sc + c] : (_Float16)0.f;
}

extern "C" void kernel_launch(void* const* d_in, const int* in_sizes, int n_in,
                              void* d_out, int out_size, void* d_ws, size_t ws_size,
                              hipStream_t stream)
{
  (void)in_sizes; (void)n_in; (void)out_size; (void)ws_size;
  const float* x    = (const float*)d_in[0];
  const float* n1g  = (const float*)d_in[1];
  const float* n1b  = (const float*)d_in[2];
  const float* qw   = (const float*)d_in[3];
  const float* kvw  = (const float*)d_in[4];
  const float* temp = (const float*)d_in[5];
  const float* qe   = (const float*)d_in[6];
  const float* rbl  = (const float*)d_in[7];
  const float* lt   = (const float*)d_in[8];
  const float* lb   = (const float*)d_in[9];
  const float* srw  = (const float*)d_in[10];
  const float* srb  = (const float*)d_in[11];
  const float* npg  = (const float*)d_in[12];
  const float* npb  = (const float*)d_in[13];
  const float* c1w  = (const float*)d_in[14];
  const float* c1b  = (const float*)d_in[15];
  const float* c2w  = (const float*)d_in[16];
  const float* c2b  = (const float*)d_in[17];
  const float* prw  = (const float*)d_in[18];
  const float* prb  = (const float*)d_in[19];
  const float* n2g  = (const float*)d_in[20];
  const float* n2b  = (const float*)d_in[21];
  const float* f1w  = (const float*)d_in[22];
  const float* f1b  = (const float*)d_in[23];
  const float* dww  = (const float*)d_in[24];
  const float* dwb  = (const float*)d_in[25];
  const float* f2w  = (const float*)d_in[26];
  const float* f2b  = (const float*)d_in[27];
  const float* tab  = (const float*)d_in[28];
  const float* sls  = (const float*)d_in[29];
  const int*   rpi  = (const int*)d_in[30];
  const unsigned char* pm = (const unsigned char*)d_in[31];
  float* outp = (float*)d_out;

  char* ws = (char*)d_ws;
  size_t off = 0;
  auto alloc = [&](size_t bytes) -> char* {
    char* p = ws + off;
    off += (bytes + 255) & ~(size_t)255;
    return p;
  };
  _Float16* h_h    = (_Float16*)alloc((size_t)TT * CC * 2);        // h f16, reused for h2
  float*    qxi    = (float*)   alloc((size_t)TT * CC * 4);        // q, later xi
  float*    kvb    = (float*)   alloc((size_t)TT * 256 * 4);       // kv, later aliased by mh
  float*    kl     = (float*)   alloc((size_t)TT * CC * 4);
  _Float16* qs_h   = (_Float16*)alloc((size_t)TT * CC * 2);
  float*    pooled = (float*)   alloc((size_t)512 * CC * 4);
  _Float16* pld_h  = (_Float16*)alloc((size_t)512 * CC * 2);
  float*    kvp    = (float*)   alloc((size_t)512 * 256 * 4);
  _Float16* kp_h   = (_Float16*)alloc((size_t)BB * HH * PL * HD * 2);
  _Float16* vpt_h  = (_Float16*)alloc((size_t)BB * HH * HD * PL * 2);
  float*    pbt    = (float*)   alloc((size_t)4 * 1024 * 4);
  _Float16* wq_h   = (_Float16*)alloc((size_t)128 * 128 * 2);
  _Float16* wkv_h  = (_Float16*)alloc((size_t)256 * 128 * 2);
  _Float16* wsr_h  = (_Float16*)alloc((size_t)128 * 128 * 2);
  _Float16* wpr_h  = (_Float16*)alloc((size_t)128 * 128 * 2);
  _Float16* wf1_h  = (_Float16*)alloc((size_t)UP * 128 * 2);       // fc1 rows padded 682->704 (zeros)
  _Float16* wf2_h  = (_Float16*)alloc((size_t)128 * HIDP * 2);     // fc2 cols padded 341->352 (zeros)
  float*    x1     = (float*)   alloc((size_t)TT * CC * 4);
  char* A0 = ws + off;                                             // attention scratch superblock
  float*    qn     = (float*)   alloc((size_t)TT * CC * 4);
  float*    AP     = (float*)   alloc((size_t)BB * HH * NN * PL * 4);
  _Float16* ap_h   = (_Float16*)alloc((size_t)BB * HH * NN * PL * 2);
  float*    xl     = (float*)   alloc((size_t)TT * CC * 4);
  _Float16* at_h   = (_Float16*)alloc((size_t)TT * CC * 2);
  float*    u      = (float*)A0;           // fc1 out [TT,UP] aliases dead attn scratch (92.27MB == block)
  _Float16* mh     = (_Float16*)kvb;       // gated MLP acts alias dead kv buffer (23MB <= 33.5MB)

  auto cvt = [&](const float* s, _Float16* d, int sr, int dr, int sc, int dc){
    long long n = (long long)dr * dc;
    cvt_k<<<dim3((unsigned)((n + 255) / 256)), 256, 0, stream>>>(s, d, sr, dr, sc, dc);
  };
  cvt(qw,  wq_h,  128, 128, 128, 128);
  cvt(kvw, wkv_h, 256, 256, 128, 128);
  cvt(srw, wsr_h, 128, 128, 128, 128);
  cvt(prw, wpr_h, 128, 128, 128, 128);
  cvt(f1w, wf1_h, HID2, UP, 128, 128);
  cvt(f2w, wf2_h, 128, 128, HID, HIDP);

  // h = LN1(x)
  ln_k<<<TT / 8, 256, 0, stream>>>(x, n1g, n1b, h_h, TT);
  // q = h @ q_w^T
  gemm_wmma<0,4><<<dim3(TT/16/8, 2, 1), 256, 0, stream>>>(h_h, wq_h, nullptr, nullptr, qxi, nullptr,
      TT, 128, 128, 128, 128, 128, 0, 0, 0, nullptr, nullptr);
  // kv = h @ kv_w^T
  gemm_wmma<0,4><<<dim3(TT/16/8, 4, 1), 256, 0, stream>>>(h_h, wkv_h, nullptr, nullptr, kvb, nullptr,
      TT, 128, 256, 128, 128, 256, 0, 0, 0, nullptr, nullptr);
  qprep_k<<<TT * HH / 8, 256, 0, stream>>>(qxi, temp, qe, sls, qn, qs_h);
  kprep_k<<<TT * HH / 8, 256, 0, stream>>>(kvb, kl);
  // xi = gelu(h @ sr_w^T + sr_b)   (overwrites q buffer)
  gemm_wmma<2,4><<<dim3(TT/16/8, 2, 1), 256, 0, stream>>>(h_h, wsr_h, srb, nullptr, qxi, nullptr,
      TT, 128, 128, 128, 128, 128, 0, 0, 0, nullptr, nullptr);
  pool_k<<<512, 128, 0, stream>>>(qxi, pooled);
  ln_k<<<512 / 8, 256, 0, stream>>>(pooled, npg, npb, pld_h, 512);
  // kvp = LN(pool) @ kv_w^T
  gemm_wmma<0,4><<<dim3(512/16/8, 4, 1), 256, 0, stream>>>(pld_h, wkv_h, nullptr, nullptr, kvp, nullptr,
      512, 128, 256, 128, 128, 256, 0, 0, 0, nullptr, nullptr);
  kpprep_k<<<512 * HH / 8, 256, 0, stream>>>(kvp, kp_h, vpt_h);
  cpb_k<<<1024, 256, 0, stream>>>(tab, c1w, c1b, c2w, c2b, pbt);
  // attn_pool[b,h] = q_s @ k_p^T + pb     (batched over 32 (b,h))
  gemm_wmma<4,4><<<dim3(NN/16/8, 1, BB*HH), 256, 0, stream>>>(qs_h, kp_h, nullptr, nullptr, AP, nullptr,
      NN, HD, PL, HD, HD, PL, (long long)NN*HD, (long long)PL*HD, (long long)NN*PL, rpi, pbt);
  attn_k<<<(BB * HH * NN) / 8, 256, 0, stream>>>(qs_h, qn, kl, kvb, AP, rbl, lt, lb, pm, xl, ap_h);
  // x_p[b,h] = a_p @ v_p, + x_l -> f16 proj input
  gemm_wmma<5,2><<<dim3(NN/16/8, 1, BB*HH), 256, 0, stream>>>(ap_h, vpt_h, nullptr, xl, nullptr, at_h,
      NN, PL, HD, PL, PL, CC, (long long)NN*PL, (long long)HD*PL, 0, nullptr, nullptr);
  // x1 = x + attn_cat @ proj_w^T + proj_b
  gemm_wmma<3,4><<<dim3(TT/16/8, 2, 1), 256, 0, stream>>>(at_h, wpr_h, prb, x, x1, nullptr,
      TT, 128, 128, 128, 128, 128, 0, 0, 0, nullptr, nullptr);
  // h2 = LN2(x1)  (reuses h buffer)
  ln_k<<<TT / 8, 256, 0, stream>>>(x1, n2g, n2b, h_h, TT);
  // u = h2 @ fc1_w^T + fc1_b   (Nd real 682, tile grid covers padded 704; ld = 704)
  gemm_wmma<1,4><<<dim3(TT/16/8, 11, 1), 256, 0, stream>>>(h_h, wf1_h, f1b, nullptr, u, nullptr,
      TT, 128, HID2, 128, 128, UP, 0, 0, 0, nullptr, nullptr);
  dwgate_k<<<TT, 352, 0, stream>>>(u, dww, dwb, mh);
  // out = x1 + mlp_act @ fc2_w^T + fc2_b   (K padded 341->352 with zeros)
  gemm_wmma<3,4><<<dim3(TT/16/8, 2, 1), 256, 0, stream>>>(mh, wf2_h, f2b, x1, outp, nullptr,
      TT, HIDP, 128, HIDP, HIDP, 128, 0, 0, 0, nullptr, nullptr);
}
